// GraphTransformerEncodingLayer_6717328851547
// MI455X (gfx1250) — compile-verified, matrix operand order checked by reading, not run on hardware
//
#include <hip/hip_runtime.h>
#include <hip/hip_fp16.h>

// ---------------- problem constants ----------------
constexpr int CB   = 4;       // batch
constexpr int CK   = 1024;    // nodes
constexpr int CL   = 32;      // seq len
constexpr int CD   = 128;     // d_model
constexpr int CDH  = 16;      // head dim
constexpr int CDFF = 512;     // ffn hidden
constexpr int CE   = 16384;   // edges
constexpr int NSEQ = CB * CK; // 4096 sequences
constexpr int LD   = CL * CD; // 4096 elems per sequence

typedef __attribute__((ext_vector_type(16))) _Float16 v16h;
typedef __attribute__((ext_vector_type(8)))  float    v8f;

// ---------------- WMMA fragment helpers (16x16x32 f16 -> f32) ----------------
// A fragment: lane holds row m = lane&15; halfs a[0..7] = K kbase+khi+0..7,
// a[8..15] = K kbase+khi+16..23 (khi = 8 for lanes 16..31) -> two b128 loads.
__device__ __forceinline__ v16h load_a_frag(const _Float16* sA, int lda,
                                            int mbase, int kbase, int lane) {
  v16h a;
  const _Float16* row = sA + (mbase + (lane & 15)) * lda + kbase + ((lane >> 4) ? 8 : 0);
#pragma unroll
  for (int v = 0; v < 8; ++v) {
    int kk = (v < 4) ? (2 * v) : (2 * v + 8);
    a[2 * v]     = row[kk];
    a[2 * v + 1] = row[kk + 1];
  }
  return a;
}

// Pre-swizzled fp16 weight fragments: [ntile][kchunk][lane][16 halfs] (32B/lane)
__device__ __forceinline__ v16h load_bh_frag(const _Float16* Bh, int KC,
                                             int nt, int kc, int lane) {
  return *(const v16h*)(Bh + ((size_t)((nt * KC + kc) * 32 + lane) << 4));
}

__device__ __forceinline__ v8f wmma_f16(v16h a, v16h b, v8f c) {
  return __builtin_amdgcn_wmma_f32_16x16x32_f16(false, a, false, b, (short)0, c,
                                                false, false);
}

// D fragment: VGPR i -> row = i + (lane>=16 ? 8 : 0), col = lane&15

// ---------------- weight pre-pack: f32 [Nout,Kin] -> swizzled fp16 fragments ----
__global__ void pack_weight(const float* __restrict__ W, _Float16* __restrict__ dst,
                            int Nout, int Kin) {
  int idx = blockIdx.x * blockDim.x + threadIdx.x;
  if (idx >= Nout * Kin) return;
  int e    = idx & 15;
  int lane = (idx >> 4) & 31;
  int tile = idx >> 9;
  int KC = Kin >> 5;
  int kc = tile % KC, nt = tile / KC;
  int v = e >> 1, p = e & 1;
  int k = ((v < 4) ? 2 * v : 2 * v + 8) + p + ((lane >> 4) ? 8 : 0) + kc * 32;
  int n = nt * 16 + (lane & 15);
  dst[idx] = (_Float16)W[n * Kin + k];   // B[k][n] = W^T fragment element
}

// ---------------- layernorm over one row of 128 ----------------
__device__ __forceinline__ void ln_row(const float* in, float* outf, _Float16* outh,
                                       const float* g, const float* bt, int row) {
  const float* r = in + row * CD;
  float m = 0.f;
#pragma unroll 8
  for (int d = 0; d < CD; ++d) m += r[d];
  m *= (1.0f / CD);
  float v = 0.f;
#pragma unroll 8
  for (int d = 0; d < CD; ++d) { float t = r[d] - m; v += t * t; }
  v *= (1.0f / CD);
  float inv = rsqrtf(v + 1e-5f);
#pragma unroll 8
  for (int d = 0; d < CD; ++d) {
    float o = (r[d] - m) * inv * g[d] + bt[d];
    outf[row * CD + d] = o;
    outh[row * CD + d] = (_Float16)o;
  }
}

// ---------------- fused transformer encoder (+ GCN weight GEMM) ----------------
// 1 block (8 wave32) per sequence. LDS = 64KB.
__global__ __launch_bounds__(256)
void encoder_kernel(const float* __restrict__ x,
                    const _Float16* __restrict__ wq_h, const float* __restrict__ b_qkv,
                    const _Float16* __restrict__ wo_h, const float* __restrict__ b_out,
                    const _Float16* __restrict__ w1_h, const float* __restrict__ b1,
                    const _Float16* __restrict__ w2_h, const float* __restrict__ b2,
                    const float* __restrict__ ln1_g, const float* __restrict__ ln1_b,
                    const float* __restrict__ ln2_g, const float* __restrict__ ln2_b,
                    const _Float16* __restrict__ wg_h, const float* __restrict__ gcn_b,
                    float* __restrict__ out, float* __restrict__ hh) {
  __shared__ float    s_res[LD];            // 16KB residual stream (f32)
  __shared__ float    s_tmp[LD];            // 16KB pre-LN scratch
  __shared__ _Float16 s_a[LD];              // 8KB  fp16 A-matrix for WMMA
  __shared__ _Float16 s_qkv[CL * 3 * CD];   // 24KB fp16 qkv; reused as ffn chunk buf

  const int tid  = threadIdx.x;
  const int wave = tid >> 5;
  const int lane = tid & 31;
  const int mhi  = (lane >> 4) ? 8 : 0;
  const int ncol = lane & 15;
  const size_t base = (size_t)blockIdx.x * LD;

  // ---- async copy x tile -> s_res (CDNA5 global_load_async_to_lds, GVS mode) ----
  {
    uint32_t lds_base = (uint32_t)(uintptr_t)(&s_res[0]);
    const float* xp = x + base;
    for (int i = tid; i < LD / 4; i += 256) {         // 1024 x 16B transfers
      uint32_t goff = (uint32_t)i * 16u;
      uint32_t loff = lds_base + (uint32_t)i * 16u;
      asm volatile("global_load_async_to_lds_b128 %0, %1, %2"
                   :: "v"(loff), "v"(goff), "s"(xp) : "memory");
    }
    asm volatile("s_wait_asynccnt 0x0" ::: "memory");
  }
  __syncthreads();
  for (int i = tid; i < LD; i += 256) s_a[i] = (_Float16)s_res[i];
  __syncthreads();

  // ---- QKV: [32,128] @ [128,384] ; 2x24 tiles over 8 waves ----
  for (int t = wave; t < 48; t += 8) {
    int mt = t / 24, nt = t % 24;
    v8f acc = {};
#pragma unroll
    for (int kc = 0; kc < 4; ++kc) {
      v16h a = load_a_frag(s_a, CD, mt * 16, kc * 32, lane);
      v16h b = load_bh_frag(wq_h, 4, nt, kc, lane);
      acc = wmma_f16(a, b, acc);
    }
    int ng = nt * 16 + ncol;
    float bias = b_qkv[ng];
#pragma unroll
    for (int i = 0; i < 8; ++i) {
      int m = mt * 16 + i + mhi;
      s_qkv[m * (3 * CD) + ng] = (_Float16)(acc[i] + bias);
    }
  }
  __syncthreads();

  // ---- attention: wave = head, lane = query row ----
  {
    const int head = wave;
    const int l = lane;
    float qv[CDH];
#pragma unroll
    for (int d = 0; d < CDH; ++d)
      qv[d] = (float)s_qkv[l * (3 * CD) + head * CDH + d];
    float sc[CL], mx = -3.0e38f;
#pragma unroll
    for (int m = 0; m < CL; ++m) {
      float s = 0.f;
#pragma unroll
      for (int d = 0; d < CDH; ++d)
        s += qv[d] * (float)s_qkv[m * (3 * CD) + CD + head * CDH + d];
      s *= 0.25f;                    // 1/sqrt(16)
      sc[m] = s;
      mx = fmaxf(mx, s);
    }
    float sum = 0.f;
#pragma unroll
    for (int m = 0; m < CL; ++m) { sc[m] = __expf(sc[m] - mx); sum += sc[m]; }
    float inv = 1.0f / sum;
#pragma unroll
    for (int d = 0; d < CDH; ++d) {
      float o = 0.f;
#pragma unroll
      for (int m = 0; m < CL; ++m)
        o += sc[m] * (float)s_qkv[m * (3 * CD) + 2 * CD + head * CDH + d];
      s_tmp[l * CD + head * CDH + d] = o * inv;
    }
  }
  __syncthreads();
  for (int i = tid; i < LD; i += 256) s_a[i] = (_Float16)s_tmp[i];
  __syncthreads();

  // ---- out-proj + residual ; 2x8 tiles, 2 per wave ----
#pragma unroll
  for (int j = 0; j < 2; ++j) {
    int t = wave * 2 + j, mt = t >> 3, nt = t & 7;
    v8f acc = {};
#pragma unroll
    for (int kc = 0; kc < 4; ++kc) {
      v16h a = load_a_frag(s_a, CD, mt * 16, kc * 32, lane);
      v16h b = load_bh_frag(wo_h, 4, nt, kc, lane);
      acc = wmma_f16(a, b, acc);
    }
    int ng = nt * 16 + ncol;
    float bias = b_out[ng];
#pragma unroll
    for (int i = 0; i < 8; ++i) {
      int m = mt * 16 + i + mhi;
      s_tmp[m * CD + ng] = acc[i] + bias + s_res[m * CD + ng];
    }
  }
  __syncthreads();
  if (tid < CL) ln_row(s_tmp, s_res, s_a, ln1_g, ln1_b, tid);  // h1
  __syncthreads();

  // ---- FFN: chunk DFF (512) into 4 x 128, keep GEMM2 accumulators live ----
  v8f acc2[2] = {{}, {}};
  _Float16* s_chunk = s_qkv;  // [32,128] fp16 relu chunk
  for (int c = 0; c < 4; ++c) {
#pragma unroll
    for (int j = 0; j < 2; ++j) {            // GEMM1 slice: h1 @ w1^T[:, c*128..]
      int t = wave * 2 + j, mt = t >> 3, nt = t & 7;
      v8f acc = {};
#pragma unroll
      for (int kc = 0; kc < 4; ++kc) {
        v16h a = load_a_frag(s_a, CD, mt * 16, kc * 32, lane);
        v16h b = load_bh_frag(w1_h, 4, c * 8 + nt, kc, lane);
        acc = wmma_f16(a, b, acc);
      }
      int ng = nt * 16 + ncol;
      float bias = b1[c * 128 + ng];
#pragma unroll
      for (int i = 0; i < 8; ++i) {
        int m = mt * 16 + i + mhi;
        s_chunk[m * CD + ng] = (_Float16)fmaxf(acc[i] + bias, 0.0f);  // relu
      }
    }
    __syncthreads();
#pragma unroll
    for (int j = 0; j < 2; ++j) {            // GEMM2 partial: chunk @ w2^T slice
      int t = wave * 2 + j, mt = t >> 3, nt = t & 7;
#pragma unroll
      for (int kc = 0; kc < 4; ++kc) {
        v16h a = load_a_frag(s_chunk, CD, mt * 16, kc * 32, lane);
        v16h b = load_bh_frag(w2_h, 16, nt, c * 4 + kc, lane);
        acc2[j] = wmma_f16(a, b, acc2[j]);
      }
    }
    __syncthreads();
  }
#pragma unroll
  for (int j = 0; j < 2; ++j) {              // finalize FFN + residual
    int t = wave * 2 + j, mt = t >> 3, nt = t & 7;
    int ng = nt * 16 + ncol;
    float bias = b2[ng];
#pragma unroll
    for (int i = 0; i < 8; ++i) {
      int m = mt * 16 + i + mhi;
      s_tmp[m * CD + ng] = acc2[j][i] + bias + s_res[m * CD + ng];
    }
  }
  __syncthreads();
  if (tid < CL) ln_row(s_tmp, s_res, s_a, ln2_g, ln2_b, tid);  // xt
  __syncthreads();

  // ---- out = xt + gcn_b (scatter kernel accumulates messages on top) ----
  for (int i = tid; i < LD; i += 256)
    out[base + i] = s_res[i] + gcn_b[i & (CD - 1)];

  // ---- hh = xt @ gcn_w^T  (GCN feature transform) ----
#pragma unroll
  for (int j = 0; j < 2; ++j) {
    int t = wave * 2 + j, mt = t >> 3, nt = t & 7;
    v8f acc = {};
#pragma unroll
    for (int kc = 0; kc < 4; ++kc) {
      v16h a = load_a_frag(s_a, CD, mt * 16, kc * 32, lane);
      v16h b = load_bh_frag(wg_h, 4, nt, kc, lane);
      acc = wmma_f16(a, b, acc);
    }
    int ng = nt * 16 + ncol;
#pragma unroll
    for (int i = 0; i < 8; ++i) {
      int m = mt * 16 + i + mhi;
      hh[base + m * CD + ng] = acc[i];
    }
  }
}

// ---------------- degree / normalization ----------------
__global__ void deg_init(float* deg) {
  int i = blockIdx.x * blockDim.x + threadIdx.x;
  if (i < CB * CK) deg[i] = 1.0f;  // self-loop weight
}
__global__ void deg_acc(const int* __restrict__ ei, const float* __restrict__ ew,
                        float* deg) {
  int i = blockIdx.x * blockDim.x + threadIdx.x;
  if (i >= CB * CE) return;
  int b = i / CE, e = i % CE;
  int col = ei[b * 2 * CE + CE + e];          // target
  unsafeAtomicAdd(&deg[b * CK + col], ew[b * CE + e]);
}
__global__ void deg_to_dinv(float* deg) {
  int i = blockIdx.x * blockDim.x + threadIdx.x;
  if (i >= CB * CK) return;
  float d = deg[i];
  deg[i] = (d > 0.0f) ? rsqrtf(d) : 0.0f;
}

// ---------------- GCN message scatter: 1 block per (edge or self-loop) ----------------
__global__ __launch_bounds__(256)
void gcn_scatter(const int* __restrict__ ei, const float* __restrict__ ew,
                 const float* __restrict__ dinv, const float* __restrict__ hh,
                 float* __restrict__ out) {
  int edge = blockIdx.x;
  int b, row, col; float w;
  if (edge < CB * CE) {
    b = edge / CE;
    int e = edge % CE;
    row = ei[b * 2 * CE + e];
    col = ei[b * 2 * CE + CE + e];
    w = ew[b * CE + e];
  } else {
    int r = edge - CB * CE;
    b = r / CK; row = col = r % CK; w = 1.0f;
  }
  float norm = dinv[b * CK + row] * w * dinv[b * CK + col];
  const float* src = hh + ((size_t)(b * CK + row)) * LD;
  float* dst = out + ((size_t)(b * CK + col)) * LD;
  for (int i = threadIdx.x; i < LD; i += 256)
    unsafeAtomicAdd(&dst[i], src[i] * norm);
}

// ---------------- launcher ----------------
extern "C" void kernel_launch(void* const* d_in, const int* in_sizes, int n_in,
                              void* d_out, int out_size, void* d_ws, size_t ws_size,
                              hipStream_t stream) {
  const float* x     = (const float*)d_in[0];
  const int*   ei    = (const int*)  d_in[1];
  const float* ew    = (const float*)d_in[2];
  const float* w_qkv = (const float*)d_in[3];
  const float* b_qkv = (const float*)d_in[4];
  const float* w_out = (const float*)d_in[5];
  const float* b_out = (const float*)d_in[6];
  const float* w1    = (const float*)d_in[7];
  const float* b1    = (const float*)d_in[8];
  const float* w2    = (const float*)d_in[9];
  const float* b2    = (const float*)d_in[10];
  const float* l1g   = (const float*)d_in[11];
  const float* l1b   = (const float*)d_in[12];
  const float* l2g   = (const float*)d_in[13];
  const float* l2b   = (const float*)d_in[14];
  const float* gcn_w = (const float*)d_in[15];
  const float* gcn_b = (const float*)d_in[16];

  float* out = (float*)d_out;
  float* hh  = (float*)d_ws;                       // [NSEQ, L, D] f32 (64MB)
  float* deg = hh + (size_t)NSEQ * LD;             // [B, K] f32
  _Float16* wq_h = (_Float16*)(deg + CB * CK);     // swizzled fp16 weights
  _Float16* wo_h = wq_h + 384 * 128;
  _Float16* w1_h = wo_h + 128 * 128;
  _Float16* w2_h = w1_h + 512 * 128;
  _Float16* wg_h = w2_h + 128 * 512;

  pack_weight<<<(384 * 128 + 255) / 256, 256, 0, stream>>>(w_qkv, wq_h, 384, 128);
  pack_weight<<<(128 * 128 + 255) / 256, 256, 0, stream>>>(w_out, wo_h, 128, 128);
  pack_weight<<<(512 * 128 + 255) / 256, 256, 0, stream>>>(w1,    w1_h, 512, 128);
  pack_weight<<<(128 * 512 + 255) / 256, 256, 0, stream>>>(w2,    w2_h, 128, 512);
  pack_weight<<<(128 * 128 + 255) / 256, 256, 0, stream>>>(gcn_w, wg_h, 128, 128);

  encoder_kernel<<<NSEQ, 256, 0, stream>>>(x, wq_h, b_qkv, wo_h, b_out,
                                           w1_h, b1, w2_h, b2, l1g, l1b, l2g, l2b,
                                           wg_h, gcn_b, out, hh);
  deg_init<<<(CB * CK + 255) / 256, 256, 0, stream>>>(deg);
  deg_acc<<<(CB * CE + 255) / 256, 256, 0, stream>>>(ei, ew, deg);
  deg_to_dinv<<<(CB * CK + 255) / 256, 256, 0, stream>>>(deg);
  gcn_scatter<<<CB * (CE + CK), 256, 0, stream>>>(ei, ew, deg, hh, out);
}
